// TransformerCell_22660247454430
// MI455X (gfx1250) — compile-verified, matrix operand order checked by reading, not run on hardware
//
#include <hip/hip_runtime.h>
#include <hip/hip_bf16.h>
#include <math.h>

// ---------------------------------------------------------------------------
// TransformerCell on gfx1250 (MI455X).
//   B=128, DIM=D_FF=1024, all fp32.
//   Roofline: writing nw [128,1024,1024] fp32 = 512MB (~22us @ 23.3TB/s) is the
//   floor; all six GEMMs total ~1.6 GFLOP (negligible). So: WMMA for GEMMs
//   (bf16x3 split => ~fp32 accuracy), exp via v_exp_f32, nw written exactly
//   once with non-temporal (streaming) stores, LDS fills via async-to-LDS.
// ---------------------------------------------------------------------------

typedef __attribute__((ext_vector_type(16))) __bf16 v16bf;
typedef __attribute__((ext_vector_type(8)))  float  v8f;
typedef __attribute__((ext_vector_type(4)))  float  v4f;

#define DIMSZ 1024
#define BSZ   128

// --- bf16 helpers via bit ops (RNE) ---
__device__ __forceinline__ __bf16 f2bf(float f) {
    unsigned u = __builtin_bit_cast(unsigned, f);
    unsigned r = (u + 0x7FFFu + ((u >> 16) & 1u)) >> 16;
    unsigned short s = (unsigned short)r;
    return __builtin_bit_cast(__bf16, s);
}
__device__ __forceinline__ float bf2f(__bf16 h) {
    unsigned short s = __builtin_bit_cast(unsigned short, h);
    unsigned u = ((unsigned)s) << 16;
    return __builtin_bit_cast(float, u);
}

// Load this lane's 16 fp32 elements of a 16x32 fragment row with explicit
// 128-bit loads, split into bf16 hi+lo.  CDNA5 16-bit A/B layout: lane owns
// K = base+{0..7} (elems 0..7) and base+16+{0..7} (elems 8..15),
// base = k0 + 8*(lane>=16).  p points at row + base (32B aligned here).
__device__ __forceinline__ void load_frag_split(const float* __restrict__ p,
                                                v16bf& hi, v16bf& lo) {
    const v4f* p4 = (const v4f*)p;
    v4f q0 = p4[0], q1 = p4[1];        // floats 0..7   -> global_load_b128 x2
    v4f q2 = p4[4], q3 = p4[5];        // floats 16..23 -> global_load_b128 x2
    float f[16];
#pragma unroll
    for (int i = 0; i < 4; ++i) {
        f[i] = q0[i]; f[4 + i] = q1[i]; f[8 + i] = q2[i]; f[12 + i] = q3[i];
    }
#pragma unroll
    for (int e = 0; e < 16; ++e) {
        __bf16 h = f2bf(f[e]);
        hi[e] = h;
        lo[e] = f2bf(f[e] - bf2f(h));
    }
}

// ---------------------------------------------------------------------------
// General GEMM:  out[M,N] = act( X[M,K] @ W[N,K]^T + bias[N] ) (+ res[M,N])
// One wave per 16x16 tile; 4 waves per 128-thread block; grid=(N/64, M/16).
// bf16x3: acc += Ahi*Bhi + Ahi*Blo + Alo*Bhi (fp32 accumulate).
// ---------------------------------------------------------------------------
__global__ __launch_bounds__(128)
void gemm_bf16x3_kernel(const float* __restrict__ X,
                        const float* __restrict__ W,
                        const float* __restrict__ bias,
                        const float* __restrict__ res,
                        float* __restrict__ out,
                        int M, int N, int K, int doRelu)
{
    const int lane = threadIdx.x & 31;
    const int wave = threadIdx.x >> 5;
    const int n0 = (blockIdx.x * 4 + wave) * 16;
    const int m0 = blockIdx.y * 16;
    const int half = lane >> 4;
    const int lr   = lane & 15;

    const float* ap = X + (size_t)(m0 + lr) * K + half * 8;
    const float* bp = W + (size_t)(n0 + lr) * K + half * 8;

    v8f acc = {};
    for (int k0 = 0; k0 < K; k0 += 32) {
        if (k0 + 32 < K) {  // global_prefetch_b8 the next K-chunk
            __builtin_prefetch(ap + k0 + 32, 0, 0);
            __builtin_prefetch(bp + k0 + 32, 0, 0);
        }
        v16bf ahi, alo, bhi, blo;
        load_frag_split(ap + k0, ahi, alo);
        load_frag_split(bp + k0, bhi, blo);
        acc = __builtin_amdgcn_wmma_f32_16x16x32_bf16(false, ahi, false, bhi,
                                                      (short)0, acc, false, false);
        acc = __builtin_amdgcn_wmma_f32_16x16x32_bf16(false, ahi, false, blo,
                                                      (short)0, acc, false, false);
        acc = __builtin_amdgcn_wmma_f32_16x16x32_bf16(false, alo, false, bhi,
                                                      (short)0, acc, false, false);
    }

    // C/D layout: lane column n = n0+lr; row m = m0 + r + 8*half.
    const int n = n0 + lr;
    const float bv = bias[n];
#pragma unroll
    for (int r = 0; r < 8; ++r) {
        const int m = m0 + r + half * 8;
        float v = acc[r] + bv;
        if (doRelu) v = v > 0.f ? v : 0.f;
        const size_t idx = (size_t)m * N + n;
        if (res) v += res[idx];
        out[idx] = v;
    }
}

// ---------------------------------------------------------------------------
// Per-batch k row min/max (closed-form softmax row max: a*kmax or a*kmin).
// ---------------------------------------------------------------------------
__global__ __launch_bounds__(256)
void kstats_kernel(const float* __restrict__ k,
                   float* __restrict__ kmax, float* __restrict__ kmin, int D)
{
    __shared__ float smax[8], smin[8];
    const int b = blockIdx.x;
    const float* kb = k + (size_t)b * D;
    float mx = -INFINITY, mn = INFINITY;
    for (int j = threadIdx.x; j < D; j += blockDim.x) {
        float v = kb[j];
        mx = fmaxf(mx, v);
        mn = fminf(mn, v);
    }
#pragma unroll
    for (int off = 16; off; off >>= 1) {  // wave32 reduction
        mx = fmaxf(mx, __shfl_xor(mx, off, 32));
        mn = fminf(mn, __shfl_xor(mn, off, 32));
    }
    const int wave = threadIdx.x >> 5, lane = threadIdx.x & 31;
    if (lane == 0) { smax[wave] = mx; smin[wave] = mn; }
    __syncthreads();
    if (threadIdx.x == 0) {
#pragma unroll
        for (int w = 1; w < 8; ++w) { mx = fmaxf(mx, smax[w]); mn = fminf(mn, smin[w]); }
        kmax[b] = mx; kmin[b] = mn;
    }
}

// ---------------------------------------------------------------------------
// Fused per-row softmax of the outer product + attention dot.
//   row (b,i): a = q[b,i]/sqrt(D);  m = a>=0 ? a*kmax : a*kmin (exact row max)
//   nw[b,i,j] = exp(a*k_j - m)/sum;  attn[b,i] = sum(e_j*v_j)/sum
// k,v rows pulled into LDS with global_load_async_to_lds_b128 (ASYNCcnt),
// pass1 accumulates (sum,dot) with wave32 shfl reductions, pass2 streams nw
// out once with non-temporal stores (512MB, no reuse, don't pollute L2).
// grid=(B, 8); block=256 (8 waves); 16 rows per wave.
// ---------------------------------------------------------------------------
__global__ __launch_bounds__(256)
void attn_softmax_kernel(const float* __restrict__ q,
                         const float* __restrict__ k,
                         const float* __restrict__ v,
                         const float* __restrict__ kmax,
                         const float* __restrict__ kmin,
                         float* __restrict__ nw,
                         float* __restrict__ attn, int D)
{
    __shared__ __align__(16) float ks[DIMSZ];
    __shared__ __align__(16) float vs[DIMSZ];
    const int b = blockIdx.x;
    const float* kb = k + (size_t)b * D;
    const float* vb = v + (size_t)b * D;

    // Async fill of LDS: 256 threads x 16B = 4KB per array (exactly D floats).
    {
        const int t = threadIdx.x;
        unsigned lk = (unsigned)(uintptr_t)(void*)&ks[t * 4];
        unsigned lv = (unsigned)(uintptr_t)(void*)&vs[t * 4];
        const float* gk = kb + t * 4;
        const float* gv = vb + t * 4;
        asm volatile("global_load_async_to_lds_b128 %0, %1, off"
                     :: "v"(lk), "v"(gk) : "memory");
        asm volatile("global_load_async_to_lds_b128 %0, %1, off"
                     :: "v"(lv), "v"(gv) : "memory");
        asm volatile("s_wait_asynccnt 0x0" ::: "memory");
    }
    __syncthreads();

    const int lane = threadIdx.x & 31;
    const int wave = threadIdx.x >> 5;
    const float scale = 1.0f / sqrtf((float)D);
    const float bmx = kmax[b], bmn = kmin[b];
    const int i0 = blockIdx.y * (D / 8);   // 128 rows per block

    for (int i = i0 + wave; i < i0 + D / 8; i += 8) {
        const float a = q[(size_t)b * D + i] * scale;
        const float m = (a >= 0.f) ? a * bmx : a * bmn;
        float sum = 0.f, dot = 0.f;
        for (int j = lane; j < D; j += 32) {
            const float e = __expf(fmaf(a, ks[j], -m));
            sum += e;
            dot += e * vs[j];
        }
#pragma unroll
        for (int off = 16; off; off >>= 1) {
            sum += __shfl_xor(sum, off, 32);
            dot += __shfl_xor(dot, off, 32);
        }
        const float rs = 1.0f / sum;
        float* __restrict__ row = nw + ((size_t)b * D + i) * D;
        for (int j = lane; j < D; j += 32) {   // coalesced streaming writes
            __builtin_nontemporal_store(__expf(fmaf(a, ks[j], -m)) * rs, &row[j]);
        }
        if (lane == 0) attn[(size_t)b * D + i] = dot * rs;
    }
}

// ---------------------------------------------------------------------------
extern "C" void kernel_launch(void* const* d_in, const int* in_sizes, int n_in,
                              void* d_out, int out_size, void* d_ws, size_t ws_size,
                              hipStream_t stream)
{
    const float* x  = (const float*)d_in[0];
    const float* Wq = (const float*)d_in[1];
    const float* bq = (const float*)d_in[2];
    const float* Wk = (const float*)d_in[3];
    const float* bk = (const float*)d_in[4];
    const float* Wv = (const float*)d_in[5];
    const float* bv = (const float*)d_in[6];
    const float* Wo = (const float*)d_in[7];
    const float* bo = (const float*)d_in[8];
    const float* W1 = (const float*)d_in[9];
    const float* b1 = (const float*)d_in[10];
    const float* W2 = (const float*)d_in[11];
    const float* b2 = (const float*)d_in[12];

    float* out = (float*)d_out;                     // [128,1024]
    float* nw  = out + (size_t)BSZ * DIMSZ;         // [128,1024,1024]

    float* ws   = (float*)d_ws;
    float* q    = ws;                               // 131072
    float* kk   = q   + (size_t)BSZ * DIMSZ;
    float* vv   = kk  + (size_t)BSZ * DIMSZ;
    float* kmx  = vv  + (size_t)BSZ * DIMSZ;        // 128
    float* kmn  = kmx + BSZ;                        // 128
    float* attn = kmn + BSZ;                        // 131072
    float* x2   = attn + (size_t)BSZ * DIMSZ;
    float* h    = x2   + (size_t)BSZ * DIMSZ;       // total ~3.1MB

    const dim3 gg(DIMSZ / 64, BSZ / 16);            // (16, 8)

    gemm_bf16x3_kernel<<<gg, 128, 0, stream>>>(x, Wq, bq, nullptr, q,  BSZ, DIMSZ, DIMSZ, 0);
    gemm_bf16x3_kernel<<<gg, 128, 0, stream>>>(x, Wk, bk, nullptr, kk, BSZ, DIMSZ, DIMSZ, 0);
    gemm_bf16x3_kernel<<<gg, 128, 0, stream>>>(x, Wv, bv, nullptr, vv, BSZ, DIMSZ, DIMSZ, 0);

    kstats_kernel<<<BSZ, 256, 0, stream>>>(kk, kmx, kmn, DIMSZ);

    attn_softmax_kernel<<<dim3(BSZ, 8), 256, 0, stream>>>(q, kk, vv, kmx, kmn,
                                                          nw, attn, DIMSZ);

    gemm_bf16x3_kernel<<<gg, 128, 0, stream>>>(attn, Wo, bo, x, x2, BSZ, DIMSZ, DIMSZ, 0);
    gemm_bf16x3_kernel<<<gg, 128, 0, stream>>>(x2, W1, b1, nullptr, h, BSZ, DIMSZ, DIMSZ, 1);
    gemm_bf16x3_kernel<<<gg, 128, 0, stream>>>(h, W2, b2, x2, out, BSZ, DIMSZ, DIMSZ, 0);
}
